// TransLayer_17746804867643
// MI455X (gfx1250) — matrix operand order repaired and checked by reading, not compiled, not run on hardware
//
#include <hip/hip_runtime.h>
#include <hip/hip_bf16.h>
#include <stdint.h>

#define N_SEQ 4096
#define DIMC  512
#define NHEAD 8
#define DH    64
#define LM    256
#define KQKV  1536

typedef __attribute__((ext_vector_type(16))) __bf16 v16bf;
typedef __attribute__((ext_vector_type(8)))  float  v8f;
typedef unsigned short u16;
typedef long long ll;

__device__ __forceinline__ u16 f2bf(float f) {
  unsigned u = __float_as_uint(f);
  unsigned r = u + 0x7FFFu + ((u >> 16) & 1u);
  return (u16)(r >> 16);
}
__device__ __forceinline__ float bf2f(u16 h) {
  return __uint_as_float(((unsigned)h) << 16);
}

// ---------------------------------------------------------------- GEMM params
struct GP {
  const u16* A; const u16* B;
  ll  sAh, sBh;
  int lda, ldb, M, N, K;
  float* C0; ll sC0h; int ldc0;          // f32 out
  u16*   O1; ll sO1h; int ldo1;          // bf16 out (primary)
  u16*   O2; ll sO2h; int ldo2;          // bf16 out (secondary)
  u16*   O3; float* F2;                  // qkv extras
  const float* bias; const float* addsrc; float* xout;  // proj epilogue
  float alpha;
  int nt;                                // nontemporal f32 stores (MODE 0)
};

// MODE 0: C0 = acc (f32), optionally nontemporal
// MODE 1: O1 = bf16(acc)
// MODE 2: qkv scatter: q(*scale)->O1, k^T->O2, v->O3(bf16)+F2(f32)
// MODE 3: O1 = bf16(alpha*I - acc)
// MODE 4: O1 = bf16(acc), O2 = bf16(7*I - acc)
// MODE 5: O1 = bf16(0.25*acc)
// MODE 6: xout = addsrc + acc + bias[col]
template<int MODE>
__global__ __launch_bounds__(256) void gemm_wmma(GP p) {
  constexpr int BM = 128, BN = 128, BK = 32, LDS_S = 40;
  // ping-pong double buffer: async tile-DMA for t+1 overlaps WMMA on t
  __shared__ __align__(16) u16 As[2][BM * LDS_S];
  __shared__ __align__(16) u16 Bt[2][BN * LDS_S];
  const int bm = blockIdx.x * BM, bn = blockIdx.y * BN, h = blockIdx.z;
  const u16* A = p.A + (ll)h * p.sAh;
  const u16* B = p.B + (ll)h * p.sBh;
  const int tid = threadIdx.x, lane = tid & 31, wave = tid >> 5;
  const int wm = wave & 1, wn = wave >> 1;          // 2x4 wave grid
  const int rlo = lane & 15, hi = lane >> 4;

  v8f acc[4][2];
#pragma unroll
  for (int m = 0; m < 4; ++m)
#pragma unroll
    for (int n = 0; n < 2; ++n)
#pragma unroll
      for (int e = 0; e < 8; ++e) acc[m][n][e] = 0.f;

  union FragU { v16bf v; uint4 q[2]; };

  // stage A tile via CDNA5 async LDS-DMA (ASYNCcnt-tracked, no VGPR bounce)
  auto stageA = [&](int k0, int buf) {
    int r = tid >> 1, kc = (tid & 1) * 16;
    const u16* src = A + (ll)(bm + r) * p.lda + (k0 + kc);
    // wave-relative LDS byte offset = low 32 bits of the generic pointer
    unsigned lds0 = (unsigned)(uintptr_t)&As[buf][r * LDS_S + kc];
    asm volatile("global_load_async_to_lds_b128 %0, %1, off\n\t"
                 "global_load_async_to_lds_b128 %0, %1, off offset:16"
                 :: "v"(lds0), "v"(src) : "memory");
    if (k0 + BK < p.K) __builtin_prefetch(src + BK, 0, 1);   // global_prefetch_b8
  };
  // stage B tile 32x128 transposed into LDS (Bt[n][k]) through VGPRs
  auto stageB = [&](int k0, int buf) {
    int kr = tid >> 3, cc = (tid & 7) * 16;
    uint4 u0, u1;
    if (bn + cc < p.N) {
      const u16* src = B + (ll)(k0 + kr) * p.ldb + (bn + cc);
      u0 = *(const uint4*)src; u1 = *(const uint4*)(src + 8);
      if (k0 + BK < p.K) __builtin_prefetch(src + (ll)BK * p.ldb, 0, 1);
    } else {
      u0.x = u0.y = u0.z = u0.w = 0u; u1 = u0;
    }
    u16 tmp[16];
    *(uint4*)&tmp[0] = u0; *(uint4*)&tmp[8] = u1;
#pragma unroll
    for (int i = 0; i < 16; ++i) Bt[buf][(cc + i) * LDS_S + kr] = tmp[i];
  };

  const int T = p.K / BK;
  stageA(0, 0);
  stageB(0, 0);
  asm volatile("s_wait_asynccnt 0x0" ::: "memory");
  __syncthreads();

  for (int t = 0; t < T; ++t) {
    const int cur = t & 1, nxt = cur ^ 1;
    if (t + 1 < T) {             // kick next tile's DMA before computing this one
      stageA((t + 1) * BK, nxt);
      stageB((t + 1) * BK, nxt);
    }

    v16bf aF[4], bF[2];
#pragma unroll
    for (int m = 0; m < 4; ++m) {       // A frag: rows wm*64+m*16..+15, K 0..31
      int row = wm * 64 + m * 16 + rlo;
      FragU f;
      f.q[0] = *(const uint4*)&As[cur][row * LDS_S + hi * 8];        // K hi*8..+7
      f.q[1] = *(const uint4*)&As[cur][row * LDS_S + hi * 8 + 16];   // K 16+hi*8..+7
      aF[m] = f.v;
    }
#pragma unroll
    for (int n = 0; n < 2; ++n) {       // B frag: cols wn*32+n*16..+15
      int col = wn * 32 + n * 16 + rlo;
      FragU f;
      f.q[0] = *(const uint4*)&Bt[cur][col * LDS_S + hi * 8];        // K hi*8..+7
      f.q[1] = *(const uint4*)&Bt[cur][col * LDS_S + hi * 8 + 16];   // K 16+hi*8..+7
      bF[n] = f.v;
    }
#pragma unroll
    for (int m = 0; m < 4; ++m)
#pragma unroll
      for (int n = 0; n < 2; ++n)
        acc[m][n] = __builtin_amdgcn_wmma_f32_16x16x32_bf16(
            false, aF[m], false, bF[n], (short)0, acc[m][n], false, false);

    // next tile's async DMA has been overlapping the WMMAs above; drain + rendezvous
    asm volatile("s_wait_asynccnt 0x0" ::: "memory");
    __syncthreads();
  }

  // epilogue (C layout: VGPR e -> M = e + 8*hi, N = rlo)
#pragma unroll
  for (int m = 0; m < 4; ++m) {
#pragma unroll
    for (int n = 0; n < 2; ++n) {
#pragma unroll
      for (int e = 0; e < 8; ++e) {
        int row = bm + wm * 64 + m * 16 + hi * 8 + e;
        int col = bn + wn * 32 + n * 16 + rlo;
        if (col >= p.N) continue;
        float val = acc[m][n][e];
        if constexpr (MODE == 0) {
          float* dst = &p.C0[(ll)h * p.sC0h + (ll)row * p.ldc0 + col];
          if (p.nt) __builtin_nontemporal_store(val, dst);  // NT: don't thrash L2
          else      *dst = val;
        } else if constexpr (MODE == 1) {
          p.O1[(ll)h * p.sO1h + (ll)row * p.ldo1 + col] = f2bf(val);
        } else if constexpr (MODE == 2) {
          int sec = col >> 9, hh = (col >> 6) & 7, d = col & 63;
          ll nd = ((ll)hh * N_SEQ + row) * DH + d;
          if (sec == 0)       p.O1[nd] = f2bf(val * 0.125f);                    // q * scale
          else if (sec == 1)  p.O2[((ll)hh * DH + d) * N_SEQ + row] = f2bf(val); // k^T
          else              { p.O3[nd] = f2bf(val); p.F2[nd] = val; }            // v
        } else if constexpr (MODE == 3) {
          float r2 = (row == col ? p.alpha : 0.f) - val;
          p.O1[(ll)h * p.sO1h + (ll)row * p.ldo1 + col] = f2bf(r2);
        } else if constexpr (MODE == 4) {
          ll o = (ll)h * p.sO1h + (ll)row * p.ldo1 + col;
          p.O1[o] = f2bf(val);
          p.O2[o] = f2bf((row == col ? 7.f : 0.f) - val);
        } else if constexpr (MODE == 5) {
          p.O1[(ll)h * p.sO1h + (ll)row * p.ldo1 + col] = f2bf(0.25f * val);
        } else if constexpr (MODE == 6) {
          ll o = (ll)row * DIMC + col;
          p.xout[o] = p.addsrc[o] + val + p.bias[col];
        }
      }
    }
  }
}

// ---------------------------------------------------------------- elementwise
__global__ __launch_bounds__(256) void f32_to_bf16_k(const float* in, u16* out, ll n) {
  ll i = (ll)blockIdx.x * 256 + threadIdx.x;
  if (i < n) out[i] = f2bf(in[i]);
}

__global__ __launch_bounds__(256) void layernorm_bf16(const float* __restrict__ x,
                                                      const float* __restrict__ g,
                                                      const float* __restrict__ b,
                                                      u16* __restrict__ out) {
  int row = blockIdx.x, tid = threadIdx.x;
  const float* xr = x + (ll)row * DIMC;
  __shared__ float red[256];
  float a = xr[tid], c = xr[tid + 256];
  red[tid] = a + c; __syncthreads();
  for (int s = 128; s > 0; s >>= 1) { if (tid < s) red[tid] += red[tid + s]; __syncthreads(); }
  float mu = red[0] * (1.0f / DIMC); __syncthreads();
  float da = a - mu, dc = c - mu;
  red[tid] = da * da + dc * dc; __syncthreads();
  for (int s = 128; s > 0; s >>= 1) { if (tid < s) red[tid] += red[tid + s]; __syncthreads(); }
  float rs = rsqrtf(red[0] * (1.0f / DIMC) + 1e-5f);
  u16* o = out + (ll)row * DIMC;
  o[tid]       = f2bf(da * rs * g[tid] + b[tid]);
  o[tid + 256] = f2bf(dc * rs * g[tid + 256] + b[tid + 256]);
}

__global__ __launch_bounds__(256) void softmax_rows(const float* __restrict__ src,
                                                    u16* __restrict__ dst, int cols) {
  ll row = blockIdx.x;
  src += row * cols; dst += row * cols;
  int tid = threadIdx.x;
  __shared__ float red[256];
  float m = -3.4e38f;
  for (int c = tid; c < cols; c += 256) m = fmaxf(m, src[c]);
  red[tid] = m; __syncthreads();
  for (int s = 128; s > 0; s >>= 1) { if (tid < s) red[tid] = fmaxf(red[tid], red[tid + s]); __syncthreads(); }
  m = red[0]; __syncthreads();
  float sum = 0.f;
  for (int c = tid; c < cols; c += 256) sum += __expf(src[c] - m);
  red[tid] = sum; __syncthreads();
  for (int s = 128; s > 0; s >>= 1) { if (tid < s) red[tid] += red[tid + s]; __syncthreads(); }
  float inv = 1.0f / red[0];
  for (int c = tid; c < cols; c += 256) dst[c] = f2bf(__expf(src[c] - m) * inv);
}

// landmarks: q[h][n][d] -> ql[h][j][d], mean of 16 consecutive n
__global__ __launch_bounds__(256) void landmark_q(const u16* __restrict__ qb, u16* __restrict__ ql) {
  ll idx = (ll)blockIdx.x * 256 + threadIdx.x;   // 8*256*64
  int d = idx & 63; ll rest = idx >> 6; int j = rest & 255; int h = (int)(rest >> 8);
  const u16* src = qb + ((ll)h * N_SEQ + (ll)j * 16) * DH + d;
  float s = 0.f;
#pragma unroll
  for (int t = 0; t < 16; ++t) s += bf2f(src[t * DH]);
  ql[idx] = f2bf(s * (1.0f / 16.f));
}

// kT[h][d][n] -> klT[h][d][j] (contiguous 16-run mean)
__global__ __launch_bounds__(256) void landmark_k(const u16* __restrict__ kT, u16* __restrict__ klT) {
  ll idx = (ll)blockIdx.x * 256 + threadIdx.x;   // 8*64*256
  int j = idx & 255; int d = (idx >> 8) & 63; int h = (int)(idx >> 14);
  const u16* src = kT + ((ll)h * DH + d) * N_SEQ + (ll)j * 16;
  float s = 0.f;
#pragma unroll
  for (int t = 0; t < 16; ++t) s += bf2f(src[t]);
  klT[idx] = f2bf(s * (1.0f / 16.f));
}

// z0 = attn2^T / max(colsum)  (rowsums of softmax == 1)
__global__ __launch_bounds__(256) void pinv_init(const u16* __restrict__ a2, u16* __restrict__ z0) {
  int h = blockIdx.x, j = threadIdx.x;
  const u16* x = a2 + (ll)h * LM * LM;
  float cs = 0.f;
  for (int i = 0; i < LM; ++i) cs += bf2f(x[i * LM + j]);
  __shared__ float red[256];
  red[j] = cs; __syncthreads();
  for (int s = 128; s > 0; s >>= 1) { if (j < s) red[j] = fmaxf(red[j], red[j + s]); __syncthreads(); }
  float inv = 1.0f / red[0];
  u16* z = z0 + (ll)h * LM * LM + (ll)j * LM;
  for (int i = 0; i < LM; ++i) z[i] = f2bf(bf2f(x[i * LM + j]) * inv);
}

// o[n][h*64+d] = bf16( outattn[h][n][d] + depthwise_conv33(v)[h][n][d] )
__global__ __launch_bounds__(256) void conv_combine(const float* __restrict__ oa,
                                                    const float* __restrict__ vf,
                                                    const float* __restrict__ wres,
                                                    u16* __restrict__ ob) {
  ll idx = (ll)blockIdx.x * 256 + threadIdx.x;   // 4096*512
  int d = idx & 63; int h = (idx >> 6) & 7; int n = (int)(idx >> 9);
  float acc = oa[((ll)h * N_SEQ + n) * DH + d];
  const float* wr = wres + h * 33;
#pragma unroll
  for (int t = 0; t < 33; ++t) {
    int mpos = n + t - 16;
    if ((unsigned)mpos < (unsigned)N_SEQ) acc += wr[t] * vf[((ll)h * N_SEQ + mpos) * DH + d];
  }
  ob[(ll)n * DIMC + h * DH + d] = f2bf(acc);
}

// ---------------------------------------------------------------- host driver
extern "C" void kernel_launch(void* const* d_in, const int* in_sizes, int n_in,
                              void* d_out, int out_size, void* d_ws, size_t ws_size,
                              hipStream_t stream) {
  const float* x    = (const float*)d_in[0];
  const float* ln_g = (const float*)d_in[1];
  const float* ln_b = (const float*)d_in[2];
  const float* wqkv = (const float*)d_in[3];
  const float* wout = (const float*)d_in[4];
  const float* bout = (const float*)d_in[5];
  const float* wres = (const float*)d_in[6];
  float* x2   = (float*)d_out;
  float* attn = (float*)d_out + (size_t)N_SEQ * DIMC;

  uintptr_t base = (uintptr_t)d_ws;
  auto alloc = [&](size_t bytes) -> void* {
    uintptr_t p = (base + 255) & ~(uintptr_t)255; base = p + bytes; return (void*)p;
  };
  const size_t SQKV = (size_t)NHEAD * N_SEQ * DH;
  const size_t SA2  = (size_t)NHEAD * LM * LM;
  const size_t SA1  = (size_t)NHEAD * N_SEQ * LM;

  u16* wqkv_b = (u16*)alloc((size_t)DIMC * KQKV * 2);
  u16* wout_b = (u16*)alloc((size_t)DIMC * DIMC * 2);
  u16* xn     = (u16*)alloc((size_t)N_SEQ * DIMC * 2);
  u16* qb     = (u16*)alloc(SQKV * 2);
  u16* kT     = (u16*)alloc(SQKV * 2);
  u16* vb     = (u16*)alloc(SQKV * 2);
  float* vf   = (float*)alloc(SQKV * 4);
  u16* ql     = (u16*)alloc((size_t)NHEAD * LM * DH * 2);
  u16* klT    = (u16*)alloc((size_t)NHEAD * DH * LM * 2);
  float* simf = (float*)alloc(SA1 * 4);
  u16* a1     = (u16*)alloc(SA1 * 2);
  u16* a2     = (u16*)alloc(SA2 * 2);
  u16* a3     = (u16*)alloc(SA1 * 2);
  u16* zA     = (u16*)alloc(SA2 * 2);
  u16* zB     = (u16*)alloc(SA2 * 2);
  u16* Pb     = (u16*)alloc(SA2 * 2);
  u16* Qb     = (u16*)alloc(SA2 * 2);
  u16* Sb     = (u16*)alloc(SA2 * 2);
  u16* Ub     = (u16*)alloc(SA2 * 2);
  u16* t1b    = (u16*)alloc((size_t)NHEAD * LM * DH * 2);
  u16* t2b    = (u16*)alloc((size_t)NHEAD * LM * DH * 2);
  u16* t3b    = (u16*)alloc(SA1 * 2);
  float* oaf  = (float*)alloc(SQKV * 4);
  u16* ob     = (u16*)alloc((size_t)N_SEQ * DIMC * 2);
  (void)ws_size; (void)in_sizes; (void)n_in; (void)out_size;

  // weights -> bf16 (deterministic, every call)
  f32_to_bf16_k<<<dim3((DIMC * KQKV + 255) / 256), 256, 0, stream>>>(wqkv, wqkv_b, (ll)DIMC * KQKV);
  f32_to_bf16_k<<<dim3((DIMC * DIMC + 255) / 256), 256, 0, stream>>>(wout, wout_b, (ll)DIMC * DIMC);

  auto run_layer = [&](const float* xin, bool pass1) {
    layernorm_bf16<<<dim3(N_SEQ), 256, 0, stream>>>(xin, ln_g, ln_b, xn);

    { GP p = {}; p.A = xn; p.B = wqkv_b; p.lda = DIMC; p.ldb = KQKV;
      p.M = N_SEQ; p.N = KQKV; p.K = DIMC;
      p.O1 = qb; p.O2 = kT; p.O3 = vb; p.F2 = vf;
      gemm_wmma<2><<<dim3(N_SEQ / 128, KQKV / 128, 1), 256, 0, stream>>>(p); }

    landmark_q<<<dim3(512), 256, 0, stream>>>(qb, ql);
    landmark_k<<<dim3(512), 256, 0, stream>>>(kT, klT);

    // sim1 = q @ k_l^T  -> softmax -> a1
    { GP p = {}; p.A = qb; p.sAh = (ll)N_SEQ * DH; p.lda = DH;
      p.B = klT; p.sBh = (ll)DH * LM; p.ldb = LM;
      p.M = N_SEQ; p.N = LM; p.K = DH;
      p.C0 = simf; p.sC0h = (ll)N_SEQ * LM; p.ldc0 = LM;
      gemm_wmma<0><<<dim3(32, 2, NHEAD), 256, 0, stream>>>(p); }
    softmax_rows<<<dim3(NHEAD * N_SEQ), 256, 0, stream>>>(simf, a1, LM);

    // sim2 = q_l @ k_l^T -> softmax -> a2
    { GP p = {}; p.A = ql; p.sAh = (ll)LM * DH; p.lda = DH;
      p.B = klT; p.sBh = (ll)DH * LM; p.ldb = LM;
      p.M = LM; p.N = LM; p.K = DH;
      p.C0 = simf; p.sC0h = (ll)LM * LM; p.ldc0 = LM;
      gemm_wmma<0><<<dim3(2, 2, NHEAD), 256, 0, stream>>>(p); }
    softmax_rows<<<dim3(NHEAD * LM), 256, 0, stream>>>(simf, a2, LM);

    // sim3 = q_l @ k^T -> softmax -> a3
    { GP p = {}; p.A = ql; p.sAh = (ll)LM * DH; p.lda = DH;
      p.B = kT; p.sBh = (ll)DH * N_SEQ; p.ldb = N_SEQ;
      p.M = LM; p.N = N_SEQ; p.K = DH;
      p.C0 = simf; p.sC0h = (ll)LM * N_SEQ; p.ldc0 = N_SEQ;
      gemm_wmma<0><<<dim3(2, 32, NHEAD), 256, 0, stream>>>(p); }
    softmax_rows<<<dim3(NHEAD * LM), 256, 0, stream>>>(simf, a3, N_SEQ);

    // Moore-Penrose pinv of a2 (6 Newton iterations), per head
    pinv_init<<<dim3(NHEAD), 256, 0, stream>>>(a2, zA);
    u16 *zc = zA, *zn = zB;
    const ll SH = (ll)LM * LM;
    for (int it = 0; it < 6; ++it) {
      { GP p = {}; p.A = a2; p.sAh = SH; p.lda = LM; p.B = zc; p.sBh = SH; p.ldb = LM;
        p.M = LM; p.N = LM; p.K = LM;
        p.O1 = Pb; p.sO1h = SH; p.ldo1 = LM; p.O2 = Qb; p.sO2h = SH; p.ldo2 = LM;
        gemm_wmma<4><<<dim3(2, 2, NHEAD), 256, 0, stream>>>(p); }       // P, Q=7I-P
      { GP p = {}; p.A = Pb; p.sAh = SH; p.lda = LM; p.B = Qb; p.sBh = SH; p.ldb = LM;
        p.M = LM; p.N = LM; p.K = LM; p.O1 = Sb; p.sO1h = SH; p.ldo1 = LM; p.alpha = 15.f;
        gemm_wmma<3><<<dim3(2, 2, NHEAD), 256, 0, stream>>>(p); }       // S=15I-P@Q
      { GP p = {}; p.A = Pb; p.sAh = SH; p.lda = LM; p.B = Sb; p.sBh = SH; p.ldb = LM;
        p.M = LM; p.N = LM; p.K = LM; p.O1 = Ub; p.sO1h = SH; p.ldo1 = LM; p.alpha = 13.f;
        gemm_wmma<3><<<dim3(2, 2, NHEAD), 256, 0, stream>>>(p); }       // U=13I-P@S
      { GP p = {}; p.A = zc; p.sAh = SH; p.lda = LM; p.B = Ub; p.sBh = SH; p.ldb = LM;
        p.M = LM; p.N = LM; p.K = LM; p.O1 = zn; p.sO1h = SH; p.ldo1 = LM;
        gemm_wmma<5><<<dim3(2, 2, NHEAD), 256, 0, stream>>>(p); }       // z=0.25*z@U
      u16* t = zc; zc = zn; zn = t;
    }

    if (pass1) {
      // t1 = attn3 @ v
      { GP p = {}; p.A = a3; p.sAh = (ll)LM * N_SEQ; p.lda = N_SEQ;
        p.B = vb; p.sBh = (ll)N_SEQ * DH; p.ldb = DH;
        p.M = LM; p.N = DH; p.K = N_SEQ;
        p.O1 = t1b; p.sO1h = (ll)LM * DH; p.ldo1 = DH;
        gemm_wmma<1><<<dim3(2, 1, NHEAD), 256, 0, stream>>>(p); }
      // t2 = z @ t1
      { GP p = {}; p.A = zc; p.sAh = SH; p.lda = LM;
        p.B = t1b; p.sBh = (ll)LM * DH; p.ldb = DH;
        p.M = LM; p.N = DH; p.K = LM;
        p.O1 = t2b; p.sO1h = (ll)LM * DH; p.ldo1 = DH;
        gemm_wmma<1><<<dim3(2, 1, NHEAD), 256, 0, stream>>>(p); }
      // out_attn = attn1 @ t2
      { GP p = {}; p.A = a1; p.sAh = (ll)N_SEQ * LM; p.lda = LM;
        p.B = t2b; p.sBh = (ll)LM * DH; p.ldb = DH;
        p.M = N_SEQ; p.N = DH; p.K = LM;
        p.C0 = oaf; p.sC0h = (ll)N_SEQ * DH; p.ldc0 = DH;
        gemm_wmma<0><<<dim3(32, 1, NHEAD), 256, 0, stream>>>(p); }
      conv_combine<<<dim3(N_SEQ * DIMC / 256), 256, 0, stream>>>(oaf, vf, wres, ob);
      // x2 = x + (o @ w_out + b_out)
      { GP p = {}; p.A = ob; p.lda = DIMC; p.B = wout_b; p.ldb = DIMC;
        p.M = N_SEQ; p.N = DIMC; p.K = DIMC;
        p.bias = bout; p.addsrc = x; p.xout = x2;
        gemm_wmma<6><<<dim3(32, 4, 1), 256, 0, stream>>>(p); }
    } else {
      // t3 = z @ attn3
      { GP p = {}; p.A = zc; p.sAh = SH; p.lda = LM;
        p.B = a3; p.sBh = (ll)LM * N_SEQ; p.ldb = N_SEQ;
        p.M = LM; p.N = N_SEQ; p.K = LM;
        p.O1 = t3b; p.sO1h = (ll)LM * N_SEQ; p.ldo1 = N_SEQ;
        gemm_wmma<1><<<dim3(2, 32, NHEAD), 256, 0, stream>>>(p); }
      // attn = attn1 @ t3   (dominant: 536 MB write, HBM-bound -> NT stores)
      { GP p = {}; p.A = a1; p.sAh = (ll)N_SEQ * LM; p.lda = LM;
        p.B = t3b; p.sBh = (ll)LM * N_SEQ; p.ldb = N_SEQ;
        p.M = N_SEQ; p.N = N_SEQ; p.K = LM;
        p.C0 = attn; p.sC0h = (ll)N_SEQ * N_SEQ; p.ldc0 = N_SEQ;
        p.nt = 1;
        gemm_wmma<0><<<dim3(32, 32, NHEAD), 256, 0, stream>>>(p); }
    }
  };

  run_layer(x, true);     // pass 1 -> x2 (in d_out)
  run_layer(x2, false);   // pass 2 -> attn (in d_out)
}
